// RopelessMLA_67095979098623
// MI455X (gfx1250) — matrix operand order repaired and checked by reading, not compiled, not run on hardware
//
#include <hip/hip_runtime.h>
#include <hip/hip_bf16.h>

#define D_    2048
#define H_    16
#define DH_   128
#define L_    512
#define B_    2
#define S_    2048
#define PAST_ 2048
#define T_    4096   // PAST + S

typedef __bf16 bf16;
typedef __attribute__((ext_vector_type(16))) __bf16 bf16x16;
typedef __attribute__((ext_vector_type(8)))  float  f32x8;

// ---------------------------------------------------------------------------
// CDNA5 async global->LDS copy (ASYNCcnt-tracked DMA, bypasses VGPRs)
// ---------------------------------------------------------------------------

__device__ __forceinline__ void async_copy_b128(const void* gsrc,
                                                void* lds_dst) {
  const unsigned ldsoff = (unsigned)(size_t)lds_dst;  // low 32b = LDS offset
  asm volatile("global_load_async_to_lds_b128 %0, %1, off"
               :: "v"(ldsoff), "v"(gsrc)
               : "memory");
}

__device__ __forceinline__ void wait_async() {
  asm volatile("s_wait_asynccnt 0x0" ::: "memory");
}

// ---------------------------------------------------------------------------
// CDNA5 WMMA fragment helpers (A and B both use contiguous k = 16*(lane/16) + element, not the ISA table order)
// ---------------------------------------------------------------------------

__device__ __forceinline__ f32x8 zerov() {
  f32x8 z;
#pragma unroll
  for (int i = 0; i < 8; ++i) z[i] = 0.f;
  return z;
}

// A-matrix 16x32 bf16: lane<16 holds row=lane, K 0-15; lane>=16 same
// rows, K 16-31. Source is row-major [16 x >=32] with leading dim ld.
__device__ __forceinline__ bf16x16 load_a_frag(const bf16* base, int ld) {
  const int lane = threadIdx.x & 31;
  const int row  = lane & 15;
  const int kg   = (lane >> 4) * 16;
  bf16x16 a;
#pragma unroll
  for (int v = 0; v < 8; ++v) {
    const int k = kg + 2 * v;
    a[2 * v]     = base[row * ld + k];
    a[2 * v + 1] = base[row * ld + k + 1];
  }
  return a;
}

// B-matrix 32x16 bf16, source row-major [K x N] (k rows), col = lane&15,
// lanes 0-15 hold K=0..15 (2 per VGPR), lanes 16-31 hold K=16..31.
__device__ __forceinline__ bf16x16 load_b_frag(const bf16* base, int ld) {
  const int lane = threadIdx.x & 31;
  const int col  = lane & 15;
  const int kg   = (lane >> 4) * 16;
  bf16x16 b;
#pragma unroll
  for (int v = 0; v < 8; ++v) {
    const int k = kg + 2 * v;
    b[2 * v]     = base[k * ld + col];
    b[2 * v + 1] = base[(k + 1) * ld + col];
  }
  return b;
}

// B fragment where the source is stored transposed: [N rows x K cols]
// (keys stored row-per-key, GEMM-K = latent dim). Per-lane reads are 16
// contiguous bf16 (2x b128) from one row; ld=520 keeps rows on distinct banks.
__device__ __forceinline__ bf16x16 load_bT_frag(const bf16* base, int ld) {
  const int lane = threadIdx.x & 31;
  const int col  = lane & 15;
  const int kg   = (lane >> 4) * 16;
  bf16x16 b;
#pragma unroll
  for (int v = 0; v < 8; ++v) {
    const int k = kg + 2 * v;
    b[2 * v]     = base[col * ld + k];
    b[2 * v + 1] = base[col * ld + k + 1];
  }
  return b;
}

__device__ __forceinline__ f32x8 wmma_bf16(bf16x16 a, bf16x16 b, f32x8 c) {
  return __builtin_amdgcn_wmma_f32_16x16x32_bf16(
      /*neg_a=*/false, a, /*neg_b=*/false, b,
      /*c_mod=*/(short)0, c, /*reuse_a=*/false, /*reuse_b=*/false);
}

// ---------------------------------------------------------------------------
// Converters
// ---------------------------------------------------------------------------

__global__ void cvt_f32_bf16(const float* __restrict__ src,
                             bf16* __restrict__ dst, size_t n) {
  size_t i = (size_t)blockIdx.x * blockDim.x + threadIdx.x;
  const size_t stride = (size_t)gridDim.x * blockDim.x;
  for (; i < n; i += stride) dst[i] = (bf16)src[i];
}

// dst[cols x rows] (bf16) = transpose(src[rows x cols]) (f32)
__global__ void transpose_f32_bf16(const float* __restrict__ src,
                                   bf16* __restrict__ dst,
                                   int rows, int cols) {
  __shared__ float t[32][33];
  const int bx = blockIdx.x * 32, by = blockIdx.y * 32;
  const int tx = threadIdx.x, ty = threadIdx.y;  // block (32, 8)
#pragma unroll
  for (int i = ty; i < 32; i += 8)
    t[i][tx] = src[(size_t)(by + i) * cols + bx + tx];
  __syncthreads();
#pragma unroll
  for (int i = ty; i < 32; i += 8)
    dst[(size_t)(bx + i) * rows + by + tx] = (bf16)t[tx][i];
}

// kv_cache [B,PAST,L] -> c_kv fp32 output rows [b*T + p] and bf16 copy
__global__ void copy_kv(const float* __restrict__ kv,
                        float* __restrict__ ckvF,
                        bf16* __restrict__ ckvB) {
  const size_t n = (size_t)B_ * PAST_ * L_;
  size_t i = (size_t)blockIdx.x * blockDim.x + threadIdx.x;
  const size_t stride = (size_t)gridDim.x * blockDim.x;
  for (; i < n; i += stride) {
    const size_t b   = i / ((size_t)PAST_ * L_);
    const size_t rem = i - b * (size_t)PAST_ * L_;
    const size_t p   = rem / L_;
    const size_t c   = rem - p * L_;
    const size_t di  = (b * T_ + p) * L_ + c;
    const float v = kv[i];
    ckvF[di] = v;
    ckvB[di] = (bf16)v;
  }
}

// LayerNorm over L=512 per row; writes fp32 rows at [b*T + PAST + s] and bf16.
__global__ __launch_bounds__(256) void layernorm_rows(
    const float* __restrict__ src, const float* __restrict__ g,
    const float* __restrict__ bta, float* __restrict__ dstF,
    bf16* __restrict__ dstB) {
  const int wid = threadIdx.x >> 5, lane = threadIdx.x & 31;
  const int row = blockIdx.x * 8 + wid;  // row in [0, B*S)
  const float* x = src + (size_t)row * L_;
  float s = 0.f, ss = 0.f;
  for (int i = lane; i < L_; i += 32) {
    const float v = x[i];
    s += v; ss += v * v;
  }
#pragma unroll
  for (int off = 16; off >= 1; off >>= 1) {
    s  += __shfl_xor(s, off, 32);
    ss += __shfl_xor(ss, off, 32);
  }
  const float mu  = s * (1.f / L_);
  const float var = ss * (1.f / L_) - mu * mu;
  const float inv = rsqrtf(var + 1e-5f);
  const int b = row / S_, srow = row % S_;
  const size_t drow = ((size_t)b * T_ + PAST_ + srow) * L_;
  for (int i = lane; i < L_; i += 32) {
    const float v = (x[i] - mu) * inv * g[i] + bta[i];
    dstF[drow + i] = v;
    dstB[drow + i] = (bf16)v;
  }
}

// ---------------------------------------------------------------------------
// Batched tiled GEMM: C[z] = A[z] * B[z], bf16 inputs, fp32 accumulate.
// Tiles 128x128x32, 256 threads = 8 waves (4x2), each wave a 32x64 patch.
// Double-buffered LDS filled by async global->LDS DMA so copy overlaps WMMA.
// ---------------------------------------------------------------------------

template <typename OutT>
__global__ __launch_bounds__(256) void gemm_bf16(
    const bf16* __restrict__ A, int zdiv, long azs1, long azs2, int lda,
    const bf16* __restrict__ Bm, long bzs1, long bzs2, int ldb,
    OutT* __restrict__ C, long czs1, long czs2, int ldc,
    int M, int N, int K) {
  __shared__ bf16 As[2][128 * 40];
  __shared__ bf16 Bs[2][32 * 144];

  const int z = blockIdx.z;
  const int z1 = z / zdiv, z2 = z % zdiv;
  const bf16* Ap = A + (size_t)z1 * azs1 + (size_t)z2 * azs2;
  const bf16* Bp = Bm + (size_t)z1 * bzs1 + (size_t)z2 * bzs2;
  OutT* Cp = C + (size_t)z1 * czs1 + (size_t)z2 * czs2;

  const int m0 = blockIdx.y * 128, n0 = blockIdx.x * 128;
  const int tid = threadIdx.x;
  const int wid = tid >> 5;
  const int wr = wid >> 1, wc = wid & 1;  // 4 x 2 wave grid

  f32x8 acc[2][4];
#pragma unroll
  for (int mf = 0; mf < 2; ++mf)
#pragma unroll
    for (int nf = 0; nf < 4; ++nf) acc[mf][nf] = zerov();

  const int ar = tid >> 1, acb = (tid & 1) * 16;   // A: 128 rows x 32
  const int br = tid >> 3, bcb = (tid & 7) * 16;   // B: 32 rows x 128

  auto stage = [&](int k0, int buf) {
    const bf16* ga = Ap + (size_t)(m0 + ar) * lda + k0 + acb;
    bf16* sa = &As[buf][ar * 40 + acb];
    async_copy_b128(ga, sa);
    async_copy_b128(ga + 8, sa + 8);
    const bf16* gb = Bp + (size_t)(k0 + br) * ldb + n0 + bcb;
    bf16* sb = &Bs[buf][br * 144 + bcb];
    async_copy_b128(gb, sb);
    async_copy_b128(gb + 8, sb + 8);
  };

  const int nk = K >> 5;
  stage(0, 0);
  for (int it = 0; it < nk; ++it) {
    const int buf = it & 1;
    wait_async();
    __syncthreads();
    if (it + 1 < nk) stage((it + 1) << 5, buf ^ 1);  // DMA overlaps compute

    bf16x16 afr[2], bfr[4];
#pragma unroll
    for (int mf = 0; mf < 2; ++mf)
      afr[mf] = load_a_frag(&As[buf][(wr * 32 + mf * 16) * 40], 40);
#pragma unroll
    for (int nf = 0; nf < 4; ++nf)
      bfr[nf] = load_b_frag(&Bs[buf][wc * 64 + nf * 16], 144);
#pragma unroll
    for (int mf = 0; mf < 2; ++mf)
#pragma unroll
      for (int nf = 0; nf < 4; ++nf)
        acc[mf][nf] = wmma_bf16(afr[mf], bfr[nf], acc[mf][nf]);
  }

  const int lane = tid & 31;
  const int col = lane & 15;
  const int rb = (lane >> 4) * 8;
#pragma unroll
  for (int mf = 0; mf < 2; ++mf)
#pragma unroll
    for (int nf = 0; nf < 4; ++nf)
#pragma unroll
      for (int r = 0; r < 8; ++r)
        Cp[(size_t)(m0 + wr * 32 + mf * 16 + rb + r) * ldc +
           n0 + wc * 64 + nf * 16 + col] = (OutT)acc[mf][nf][r];
}

// ---------------------------------------------------------------------------
// Flash attention: scores = q_lat @ c_kv^T (K=L=512), causal(offset=PAST)
// softmax, ctx = P @ v. Block = 8 waves x 16 q-rows; Q in registers as 16
// A-frags; K/V double-buffered in LDS via async DMA (overlaps WMMA stream).
// ---------------------------------------------------------------------------

__global__ __launch_bounds__(256) void flash_attn(
    const bf16* __restrict__ qlat,  // [B,H,S,L]
    const bf16* __restrict__ ckv,   // [B*T, L]
    const bf16* __restrict__ vmat,  // [B*T, D] (heads in columns)
    bf16* __restrict__ ctx) {       // [B*S, D]
  __shared__ bf16 Ks[2][32 * 520];    // keys: [key][latent], 2 x 33.3 KB
  __shared__ bf16 Vs[2][32 * 136];    // vals: [key][dh],     2 x  8.7 KB
  __shared__ bf16 Ps[8 * 16 * 40];    // per-wave P staging,      10.2 KB

  const int qb = blockIdx.x * 128;
  const int h  = blockIdx.y;
  const int b  = blockIdx.z;
  const int tid = threadIdx.x;
  const int w = tid >> 5;
  const int lane = tid & 31;
  const int colid = lane & 15;
  const int rbase = (lane >> 4) * 8;

  // Preload this wave's 16 Q rows as 16 A-frags (L = 16 * 32).
  const bf16* qbase = qlat + ((size_t)(b * H_ + h) * S_ + qb + w * 16) * L_;
  bf16x16 qf[16];
#pragma unroll
  for (int ks = 0; ks < 16; ++ks) qf[ks] = load_a_frag(qbase + ks * 32, L_);

  float m[8], l[8];
  f32x8 oacc[8];
#pragma unroll
  for (int r = 0; r < 8; ++r) { m[r] = -1e30f; l[r] = 0.f; }
#pragma unroll
  for (int n = 0; n < 8; ++n) oacc[n] = zerov();

  const float scale = 0.08838834764831845f;  // 1/sqrt(DH)
  const int ntiles = ((qb + 127 + PAST_) >> 5) + 1;

  const int kr = tid >> 3;
  const int kcb = (tid & 7) * 64;
  const int vcb = (tid & 7) * 16;

  auto stageKV = [&](int kb, int buf) {
    const bf16* gk = ckv + (size_t)(b * T_ + kb + kr) * L_ + kcb;
    bf16* sk = &Ks[buf][kr * 520 + kcb];
#pragma unroll
    for (int i = 0; i < 8; ++i) async_copy_b128(gk + i * 8, sk + i * 8);
    const bf16* gv = vmat + (size_t)(b * T_ + kb + kr) * D_ + h * DH_ + vcb;
    bf16* sv = &Vs[buf][kr * 136 + vcb];
    async_copy_b128(gv, sv);
    async_copy_b128(gv + 8, sv + 8);
  };

  stageKV(0, 0);
  for (int nt = 0; nt < ntiles; ++nt) {
    const int kb = nt * 32;
    const int buf = nt & 1;
    wait_async();
    __syncthreads();
    if (nt + 1 < ntiles) stageKV(kb + 32, buf ^ 1);  // DMA overlaps compute

    // scores: 2 n-frags (2 x 16 keys), K-dim = 512
    f32x8 sacc[2];
#pragma unroll
    for (int nf = 0; nf < 2; ++nf) {
      sacc[nf] = zerov();
#pragma unroll
      for (int ks = 0; ks < 16; ++ks) {
        bf16x16 kf = load_bT_frag(&Ks[buf][(nf * 16) * 520 + ks * 32], 520);
        sacc[nf] = wmma_bf16(qf[ks], kf, sacc[nf]);
      }
    }

    const bool needMask = (kb + 31) > (qb + PAST_);
#pragma unroll
    for (int r = 0; r < 8; ++r) {
      const int irow = qb + w * 16 + rbase + r;
      float p0 = sacc[0][r] * scale;
      float p1 = sacc[1][r] * scale;
      if (needMask) {
        if (kb + colid > irow + PAST_) p0 = -1e30f;
        if (kb + 16 + colid > irow + PAST_) p1 = -1e30f;
      }
      float tm = fmaxf(p0, p1);
#pragma unroll
      for (int off = 8; off >= 1; off >>= 1)
        tm = fmaxf(tm, __shfl_xor(tm, off, 32));
      const float nm = fmaxf(m[r], tm);
      const float al = __expf(m[r] - nm);
      m[r] = nm;
      p0 = __expf(p0 - nm);
      p1 = __expf(p1 - nm);
      float rs = p0 + p1;
#pragma unroll
      for (int off = 8; off >= 1; off >>= 1) rs += __shfl_xor(rs, off, 32);
      l[r] = l[r] * al + rs;
#pragma unroll
      for (int n = 0; n < 8; ++n) oacc[n][r] *= al;
      Ps[(w * 16 + rbase + r) * 40 + colid] = (bf16)p0;
      Ps[(w * 16 + rbase + r) * 40 + 16 + colid] = (bf16)p1;
    }

    // ctx += P(16x32) @ V(32x128); per-wave LDS slab, same-wave dep only.
    bf16x16 pa = load_a_frag(&Ps[w * 16 * 40], 40);
#pragma unroll
    for (int n = 0; n < 8; ++n) {
      bf16x16 vf = load_b_frag(&Vs[buf][n * 16], 136);
      oacc[n] = wmma_bf16(pa, vf, oacc[n]);
    }
    __syncthreads();  // all readers of Ks/Vs[buf] done before tile nt+2 stage
  }

#pragma unroll
  for (int n = 0; n < 8; ++n)
#pragma unroll
    for (int r = 0; r < 8; ++r) {
      const float o = oacc[n][r] / l[r];
      const size_t row = (size_t)b * S_ + qb + w * 16 + rbase + r;
      ctx[row * D_ + h * DH_ + n * 16 + colid] = (bf16)o;
    }
}

// ---------------------------------------------------------------------------
// Host orchestration
// ---------------------------------------------------------------------------

extern "C" void kernel_launch(void* const* d_in, const int* in_sizes, int n_in,
                              void* d_out, int out_size, void* d_ws,
                              size_t ws_size, hipStream_t stream) {
  const float* x    = (const float*)d_in[0];
  const float* kv   = (const float*)d_in[1];
  const float* Wq   = (const float*)d_in[2];
  const float* Wdkv = (const float*)d_in[3];
  const float* Wuk  = (const float*)d_in[4];
  const float* Wuv  = (const float*)d_in[5];
  const float* Wo   = (const float*)d_in[6];
  const float* lng  = (const float*)d_in[7];
  const float* lnb  = (const float*)d_in[8];

  float* outF = (float*)d_out;                    // [B*S, D]
  float* ckvF = outF + (size_t)B_ * S_ * D_;      // [B, T, L]

  char* ws = (char*)d_ws;
  size_t off = 0;
  auto alloc = [&](size_t bytes) -> void* {
    void* p = ws + off;
    off = (off + bytes + 255) & ~(size_t)255;
    return p;
  };
  bf16* x_bf     = (bf16*)alloc((size_t)B_ * S_ * D_ * 2);
  bf16* Wq_bf    = (bf16*)alloc((size_t)D_ * D_ * 2);
  bf16* Wuk_bf   = (bf16*)alloc((size_t)D_ * L_ * 2);
  bf16* WdkvT_bf = (bf16*)alloc((size_t)D_ * L_ * 2);   // [D, L]
  bf16* WuvT_bf  = (bf16*)alloc((size_t)L_ * D_ * 2);   // [L, D]
  bf16* WoT_bf   = (bf16*)alloc((size_t)D_ * D_ * 2);   // [D, D]
  bf16* abs_bf   = (bf16*)alloc((size_t)D_ * L_ * 2);   // absorbed [H*DH, L]
  float* newc_f  = (float*)alloc((size_t)B_ * S_ * L_ * 4);
  bf16* ckv_bf   = (bf16*)alloc((size_t)B_ * T_ * L_ * 2);
  bf16* v_bf     = (bf16*)alloc((size_t)B_ * T_ * D_ * 2);
  bf16* qlat_bf  = (bf16*)alloc((size_t)B_ * H_ * S_ * L_ * 2);
  bf16* ctx_bf   = (bf16*)alloc((size_t)B_ * S_ * D_ * 2);

  // 1) bf16 conversions of row-major operands
  cvt_f32_bf16<<<2048, 256, 0, stream>>>(x, x_bf, (size_t)B_ * S_ * D_);
  cvt_f32_bf16<<<2048, 256, 0, stream>>>(Wq, Wq_bf, (size_t)D_ * D_);
  cvt_f32_bf16<<<512, 256, 0, stream>>>(Wuk, Wuk_bf, (size_t)D_ * L_);

  // 2) transposed bf16 weights ([K, N] layout for x @ W^T style GEMMs)
  transpose_f32_bf16<<<dim3(D_ / 32, L_ / 32), dim3(32, 8), 0, stream>>>(
      Wdkv, WdkvT_bf, L_, D_);  // [L,D] -> [D,L]
  transpose_f32_bf16<<<dim3(L_ / 32, D_ / 32), dim3(32, 8), 0, stream>>>(
      Wuv, WuvT_bf, D_, L_);    // [D,L] -> [L,D]
  transpose_f32_bf16<<<dim3(D_ / 32, D_ / 32), dim3(32, 8), 0, stream>>>(
      Wo, WoT_bf, D_, D_);      // [D,D] -> [D,D]^T

  // 3) absorbed = W_q @ W_uk : [2048,2048]x[2048,512]
  gemm_bf16<bf16><<<dim3(L_ / 128, D_ / 128, 1), 256, 0, stream>>>(
      Wq_bf, 1, 0, 0, D_, Wuk_bf, 0, 0, L_, abs_bf, 0, 0, L_, D_, L_, D_);

  // 4) new_c (pre-LN) = x @ W_dkv^T : [4096,2048]x[2048,512]
  gemm_bf16<float><<<dim3(L_ / 128, (B_ * S_) / 128, 1), 256, 0, stream>>>(
      x_bf, 1, 0, 0, D_, WdkvT_bf, 0, 0, L_, newc_f, 0, 0, L_,
      B_ * S_, L_, D_);

  // 5) c_kv assembly: cached part + layernormed new part (fp32 out + bf16 ws)
  copy_kv<<<2048, 256, 0, stream>>>(kv, ckvF, ckv_bf);
  layernorm_rows<<<(B_ * S_) / 8, 256, 0, stream>>>(newc_f, lng, lnb, ckvF,
                                                    ckv_bf);

  // 6) v = c_kv @ W_uv^T : [8192,512]x[512,2048]
  gemm_bf16<bf16><<<dim3(D_ / 128, (B_ * T_) / 128, 1), 256, 0, stream>>>(
      ckv_bf, 1, 0, 0, L_, WuvT_bf, 0, 0, D_, v_bf, 0, 0, D_,
      B_ * T_, D_, L_);

  // 7) q_lat[b,h] = q[b,:,h,:] @ absorbed[h] : 32 x ([2048,128]x[128,512])
  //    z = b*H + h ; z1 = b (azs1 = S*D), z2 = h (azs2 = DH)
  gemm_bf16<bf16><<<dim3(L_ / 128, S_ / 128, B_ * H_), 256, 0, stream>>>(
      x_bf, H_, (long)S_ * D_, (long)DH_, D_,
      abs_bf, 0, (long)DH_ * L_, L_,
      qlat_bf, (long)H_ * S_ * L_, (long)S_ * L_, L_,
      S_, L_, DH_);

  // 8) flash attention -> ctx (bf16, [B*S, D] with head-blocked columns)
  flash_attn<<<dim3(S_ / 128, H_, B_), 256, 0, stream>>>(qlat_bf, ckv_bf,
                                                         v_bf, ctx_bf);

  // 9) out = ctx @ W_o^T : [4096,2048]x[2048,2048] -> fp32 d_out
  gemm_bf16<float><<<dim3(D_ / 128, (B_ * S_) / 128, 1), 256, 0, stream>>>(
      ctx_bf, 1, 0, 0, D_, WoT_bf, 0, 0, D_, outF, 0, 0, D_,
      B_ * S_, D_, D_);

  (void)in_sizes; (void)n_in; (void)out_size; (void)ws_size;
}
